// CGPODE_62483184222730
// MI455X (gfx1250) — matrix operand order repaired and checked by reading, not compiled, not run on hardware
//
#include <hip/hip_runtime.h>
#include <hip/hip_bf16.h>

// CDNA5 / gfx1250 implementation.
//
// out[n,o,v,l] = sum_k sum_c W[o, k*64+c] * ((I + 0.125*A)^k x)[n,c,v,l] + b[o]
//
// Pipeline (all on `stream`, deterministic, 20 launches):
//   prep:  x(f32)[n,c,w,l] -> bf16 state buf0 laid out [n,c,l,w_pad512] (pad=0)
//          M = bf16(I + 0.125*adj), padded 512x512 (pad rows/cols = 0)
//          W -> bf16
//   k = 0..8:
//     chan_acc : out(+)= W_k · state_k          (v_wmma_f32_16x16x32_bf16)
//     node_step: state_{k+1} = M · state_k      (v_wmma_f32_16x16x32_bf16), k<8
//
// node_step batches 4 (n,c) blocks per 512-thread workgroup so every M-tile
// (A fragment) loaded from L2 feeds 4 WMMAs -> 4x less A-fragment L2 traffic
// (the binding resource at the bf16 WMMA roofline). State layout [l][w512]
// keeps staging a straight b128 copy, output stores packed b128, and the zero
// K-padding self-maintaining (M pad rows are zero).
// Workspace: 512KB (M) + 72KB (W) + 2 x 50.3MB states (~101 MB).

typedef unsigned int   u32;
typedef unsigned short u16;
typedef __attribute__((ext_vector_type(16))) __bf16 v16bf;
typedef __attribute__((ext_vector_type(8)))  float  v8f;
typedef __attribute__((ext_vector_type(8)))  u32    v8u;

#define NFE   8
#define STEPF 0.125f
#define NB    64        // batch
#define CH    64        // channels
#define NV    500       // nodes
#define NL    12        // length
#define VL    6000      // NV*NL (output col count)
#define VP    512       // padded node dim
#define LWP   6144      // NL*VP (state cols per (n,c))
#define LDSW  520       // node kernel: padded w-stride (halves); 1040B row pitch
                        //   -> 16 column-lanes land on banks {0,4,...,60}: no conflict
#define LDSB  (16 * LDSW)  // one (n,c) LDS image (halves)
#define NCB   4         // (n,c) blocks batched per node workgroup
#define CST   72        // chan kernel: padded c-stride (halves)
#define CGRP  48        // 6144 / 128 column groups per n (exact)

__device__ __forceinline__ u16 f2bf(float f) {
  u32 u = __float_as_uint(f);
  u += 0x7FFFu + ((u >> 16) & 1u);   // round-to-nearest-even
  return (u16)(u >> 16);
}
__device__ __forceinline__ u32 pk2bf(float lo, float hi) {
  return (u32)f2bf(lo) | ((u32)f2bf(hi) << 16);
}

// ---------------- prep kernels ----------------

// x [n,c,w(500),l(12)] f32  ->  buf0 [n,c,l(12),w(512)] bf16, pad w>=500 zeroed
__global__ __launch_bounds__(256) void cvt_x_kernel(const float* __restrict__ x,
                                                    u16* __restrict__ dst) {
  const int i = blockIdx.x * 256 + threadIdx.x;   // over NB*CH*NL*VP
  const int w  = i & (VP - 1);
  const int l  = (i >> 9) % NL;
  const int nc = i / LWP;
  float v = 0.0f;
  if (w < NV) v = x[(nc * NV + w) * NL + l];
  dst[i] = f2bf(v);
}

__global__ __launch_bounds__(256) void build_M_kernel(const float* __restrict__ adj,
                                                      u16* __restrict__ Mbf) {
  const int i = blockIdx.x * 256 + threadIdx.x;   // covers 512*512
  const int v = i >> 9, w = i & (VP - 1);
  float m = 0.0f;
  if (v < NV && w < NV) m = ((v == w) ? 1.0f : 0.0f) + STEPF * adj[v * NV + w];
  Mbf[i] = f2bf(m);
}

__global__ __launch_bounds__(256) void build_W_kernel(const float* __restrict__ W,
                                                      u16* __restrict__ Wbf, int n) {
  const int i = blockIdx.x * 256 + threadIdx.x;
  if (i < n) Wbf[i] = f2bf(W[i]);
}

// ------- node step: for 4 consecutive (n,c): y_out = M (512x512) * y_in -------
// 512 threads = 16 waves. Wave wv owns v-tiles {wv, wv+16} and all 4 nc-slots.
// Per K-chunk: build 4 B fragments (2 ds_load_b128 each, from 520-stride LDS
// images), then per v-tile load one A fragment (2 global b128 from L2-resident
// M) and issue 4 WMMAs with it.

__global__ __launch_bounds__(512) void node_step_kernel(const u16* __restrict__ yin,
                                                        u16* __restrict__ yout,
                                                        const u16* __restrict__ Mbf) {
  __shared__ u16 sX[NCB * LDSB];
  const int tid  = threadIdx.x;
  const int base = blockIdx.x * (NCB * LWP);        // 4 consecutive (n,c)

  // straight copy: per nc-slot 12 rows x 512 halves = 768 x 16B chunks
  for (int q = tid; q < NCB * (NL * VP) / 8; q += 512) {
    const int b   = q / 768;
    const int rem = q - b * 768;
    const int l   = rem >> 6;                       // 64 chunks per row
    const int o8  = (rem & 63) * 8;
    *(uint4*)(&sX[b * LDSB + l * LDSW + o8]) =
        *(const uint4*)(&yin[base + b * LWP + l * VP + o8]);
  }
  // zero LDS rows 12..15 of each slot (feed discarded B columns 12..15)
  for (int i = tid; i < NCB * 4 * LDSW; i += 512) {
    const int b = i / (4 * LDSW);
    sX[b * LDSB + NL * LDSW + (i - b * 4 * LDSW)] = 0;
  }
  __syncthreads();

  const int lane  = tid & 31;
  const int wv    = tid >> 5;        // wave id 0..15
  const int col   = lane & 15;       // B/D column (l) and A row-in-tile (v)
  const int khalf = lane >> 4;       // fragment half select

  const v8f vz = {0.f,0.f,0.f,0.f,0.f,0.f,0.f,0.f};
  v8f acc[2][NCB];
  #pragma unroll
  for (int t = 0; t < 2; ++t)
    #pragma unroll
    for (int b = 0; b < NCB; ++b) acc[t][b] = vz;

  for (int kc = 0; kc < 16; ++kc) {                 // K = 512 in chunks of 32
    const int wb = kc * 32 + khalf * 8;
    // 4 B fragments (one per nc-slot); VGPR j packs K = (wb + 2j + (j>=4 ? 8 : 0), +1)
    v16bf bfrag[NCB];
    #pragma unroll
    for (int b = 0; b < NCB; ++b) {
      v8u bu;
      #pragma unroll
      for (int j = 0; j < 8; ++j)
        bu[j] = *(const u32*)(&sX[b * LDSB + col * LDSW + wb + 2 * j + (j >= 4 ? 8 : 0)]);
      bfrag[b] = __builtin_bit_cast(v16bf, bu);
    }

    #pragma unroll
    for (int t = 0; t < 2; ++t) {
      const int vt = wv + t * 16;                   // v-tile 0..31
      const u16* ap = Mbf + (vt * 16 + col) * VP + kc * 32 + khalf * 8;
      uint4 a0 = *(const uint4*)(ap);
      uint4 a1 = *(const uint4*)(ap + 16);
      v8u au;
      au[0] = a0.x; au[1] = a0.y; au[2] = a0.z; au[3] = a0.w;
      au[4] = a1.x; au[5] = a1.y; au[6] = a1.z; au[7] = a1.w;
      const v16bf afrag = __builtin_bit_cast(v16bf, au);
      #pragma unroll
      for (int b = 0; b < NCB; ++b)                 // A reused 4x
        acc[t][b] = __builtin_amdgcn_wmma_f32_16x16x32_bf16(
            false, afrag, false, bfrag[b], (short)0, acc[t][b], false, false);
    }
  }

  // packed stores: 8 f32 -> 4 dwords bf16 -> one b128 per (tile, nc-slot)
  if (col < NL) {
    #pragma unroll
    for (int t = 0; t < 2; ++t) {
      const int v0 = (wv + t * 16) * 16 + khalf * 8;
      #pragma unroll
      for (int b = 0; b < NCB; ++b) {
        uint4 pk;
        pk.x = pk2bf(acc[t][b][0], acc[t][b][1]);
        pk.y = pk2bf(acc[t][b][2], acc[t][b][3]);
        pk.z = pk2bf(acc[t][b][4], acc[t][b][5]);
        pk.w = pk2bf(acc[t][b][6], acc[t][b][7]);
        *(uint4*)(&yout[base + b * LWP + col * VP + v0]) = pk;
      }
    }
  }
}

// ------- channel accumulate: out[n,o,:] (+)= W_k (64x64) * y_k[n,:,:] -------
// One workgroup per (n, 128-column group) over 6144 padded columns. Wave i owns
// column-tile i and all 4 o-tiles; K = 64 = two chunks of 32. y staged
// transposed in LDS [col][c] (stride 72 -> conflict-free fragment reads).

__global__ __launch_bounds__(256) void chan_acc_kernel(const u16* __restrict__ ybf,
                                                       const u16* __restrict__ Wbf,
                                                       const float* __restrict__ bias,
                                                       float* __restrict__ out,
                                                       int kIdx, int accumulate) {
  __shared__ u16 sY[128 * CST];
  const int tid = threadIdx.x;
  const int n   = blockIdx.x / CGRP;
  const int cg  = blockIdx.x % CGRP;
  const int cb  = cg * 128;

  for (int i = tid; i < 128 * CH; i += 256) {
    const int c    = i >> 7;
    const int colI = i & 127;
    sY[colI * CST + c] = ybf[(n * CH + c) * LWP + cb + colI];
  }
  __syncthreads();

  const int lane  = tid & 31;
  const int wv    = tid >> 5;
  const int col   = lane & 15;
  const int khalf = lane >> 4;

  const v8f vz = {0.f,0.f,0.f,0.f,0.f,0.f,0.f,0.f};
  v8f acc[4];
  #pragma unroll
  for (int t = 0; t < 4; ++t) acc[t] = vz;

  #pragma unroll
  for (int kc = 0; kc < 2; ++kc) {
    v8u bu;
    const int cbase = kc * 32 + khalf * 8;
    #pragma unroll
    for (int j = 0; j < 8; ++j)
      bu[j] = *(const u32*)(&sY[(wv * 16 + col) * CST + cbase + 2 * j + (j >= 4 ? 8 : 0)]);
    const v16bf bfrag = __builtin_bit_cast(v16bf, bu);

    #pragma unroll
    for (int t = 0; t < 4; ++t) {
      const u16* ap = Wbf + (t * 16 + col) * ((NFE + 1) * CH)
                          + kIdx * CH + kc * 32 + khalf * 8;
      uint4 a0 = *(const uint4*)(ap);
      uint4 a1 = *(const uint4*)(ap + 16);
      v8u au;
      au[0] = a0.x; au[1] = a0.y; au[2] = a0.z; au[3] = a0.w;
      au[4] = a1.x; au[5] = a1.y; au[6] = a1.z; au[7] = a1.w;
      acc[t] = __builtin_amdgcn_wmma_f32_16x16x32_bf16(
          false, __builtin_bit_cast(v16bf, au),
          false, bfrag, (short)0, acc[t], false, false);
    }
  }

  // padded col (l*512 + w) -> reference index (v*12 + l); drop pad w >= 500
  const int gcp = cb + wv * 16 + col;
  const int l   = gcp >> 9;
  const int w   = gcp & (VP - 1);
  if (w < NV) {
    const int vl = w * NL + l;
    #pragma unroll
    for (int t = 0; t < 4; ++t) {
      #pragma unroll
      for (int r = 0; r < 8; ++r) {
        const int o    = t * 16 + khalf * 8 + r;
        const int addr = (n * CH + o) * VL + vl;
        const float v  = acc[t][r];
        if (accumulate) out[addr] += v;
        else            out[addr]  = v + bias[o];
      }
    }
  }
}

// ---------------- host launch ----------------

extern "C" void kernel_launch(void* const* d_in, const int* in_sizes, int n_in,
                              void* d_out, int out_size, void* d_ws, size_t ws_size,
                              hipStream_t stream) {
  const float* x    = (const float*)d_in[0];
  const float* adj  = (const float*)d_in[1];
  const float* W    = (const float*)d_in[2];
  const float* bias = (const float*)d_in[3];
  float* out = (float*)d_out;

  char* ws = (char*)d_ws;
  size_t off = 0;
  auto alloc = [&](size_t bytes) -> char* {
    char* p = ws + off;
    off += (bytes + 255) & ~(size_t)255;
    return p;
  };
  u16* Mbf  = (u16*)alloc((size_t)VP * VP * 2);                 // 512 KB
  u16* Wbf  = (u16*)alloc((size_t)CH * (NFE + 1) * CH * 2);     // 72 KB
  u16* buf0 = (u16*)alloc((size_t)NB * CH * LWP * 2);           // 50.3 MB
  u16* buf1 = (u16*)alloc((size_t)NB * CH * LWP * 2);           // 50.3 MB

  const int xPadCount = NB * CH * LWP;   // 25,165,824
  cvt_x_kernel<<<xPadCount / 256, 256, 0, stream>>>(x, buf0);
  build_M_kernel<<<(VP * VP) / 256, 256, 0, stream>>>(adj, Mbf);
  build_W_kernel<<<(CH * (NFE + 1) * CH + 255) / 256, 256, 0, stream>>>(
      W, Wbf, CH * (NFE + 1) * CH);

  u16* bufs[2] = { buf0, buf1 };
  for (int k = 0; k <= NFE; ++k) {
    chan_acc_kernel<<<NB * CGRP, 256, 0, stream>>>(
        bufs[k & 1], Wbf, bias, out, k, (k > 0) ? 1 : 0);
    if (k < NFE)
      node_step_kernel<<<(NB * CH) / NCB, 512, 0, stream>>>(
          bufs[k & 1], bufs[(k + 1) & 1], Mbf);
  }
}
